// OrthoLinear_4183298146442
// MI455X (gfx1250) — compile-verified, matrix operand order checked by reading, not run on hardware
//
#include <hip/hip_runtime.h>

// ---------------- problem constants (match reference) ----------------
#define BATCH   4
#define SEQ     2048
#define M_TOT   (BATCH * SEQ)   // 8192
#define KDIM    4096            // IN_F
#define NDIM    4096            // OUT_F
#define K_OUTL  204
#define ALPHA_F 1.0f

// ---------------- vector types ----------------
typedef __attribute__((ext_vector_type(16))) __bf16          v16bf;
typedef __attribute__((ext_vector_type(8)))  float           v8f;
typedef __attribute__((ext_vector_type(8)))  unsigned short  u16x8;
typedef __attribute__((ext_vector_type(4)))  unsigned short  u16x4;
typedef __attribute__((ext_vector_type(4)))  float           f32x4;

union FragBF {
    u16x8 h[2];   // two contiguous 8-element bf16 chunks per lane
    v16bf v;      // 16 bf16 = 8 VGPRs, matches WMMA 16-bit A/B operand
};

__device__ __forceinline__ unsigned short f2bf(float f) {
    __bf16 b = (__bf16)f;                       // native v_cvt path on gfx1250
    return __builtin_bit_cast(unsigned short, b);
}

// =====================================================================
// Phase 1: build W_bf16[o][i] = (int4 - 8)*scale + ALPHA*scatter(outliers)
// One workgroup per output row o.
// =====================================================================
__global__ __launch_bounds__(256)
void ortho_dequant_kernel(const int* __restrict__ base_packed,
                          const float* __restrict__ scales,
                          const _Float16* __restrict__ ortho_vals,
                          const int* __restrict__ ortho_idx,
                          unsigned short* __restrict__ Wout) {
    __shared__ float row[KDIM];               // 16 KB
    const int o   = blockIdx.x;
    const int tid = threadIdx.x;
    const float scale = scales[o];

    const int* bp = base_packed + (size_t)o * (KDIM / 2);
    for (int i = tid; i < KDIM / 2; i += 256) {
        int v = bp[i];
        row[2 * i + 0] = (float)((v & 0xF) - 8) * scale;
        row[2 * i + 1] = (float)(((v >> 4) & 0xF) - 8) * scale;
    }
    __syncthreads();

    // scatter sparse residual (duplicate indices accumulate, like .at[].add)
    for (int j = tid; j < K_OUTL; j += 256) {
        int   idx = ortho_idx[(size_t)o * K_OUTL + j];
        float val = (float)ortho_vals[(size_t)o * K_OUTL + j] * ALPHA_F;
        atomicAdd(&row[idx], val);
    }
    __syncthreads();

    unsigned short* wr = Wout + (size_t)o * KDIM;
    for (int i = tid; i < KDIM; i += 256)
        wr[i] = f2bf(row[i]);
}

// =====================================================================
// Phase 2: out[M,N] = X[M,K] * W[N,K]^T  via v_wmma_f32_16x16x32_bf16
// Block tile 256(M) x 128(N), BK=64, 256 threads = 8 waves (wave32).
// 4 waves along M x 2 along N; each wave owns 64x64 = 4x4 WMMA tiles
// -> 16 WMMAs per 16 ds_load_b128 (1.0 LDS loads per WMMA).
// =====================================================================
#define BM 256
#define BN 128
#define BK 64
#define LDS_PITCH (BK + 8)   // +8 bf16 pad -> conflict-free 16-row column reads

__global__ __launch_bounds__(256)
void ortho_gemm_kernel(const float* __restrict__ X,
                       const unsigned short* __restrict__ W,
                       float* __restrict__ out) {
    __shared__ unsigned short A_sh[BM][LDS_PITCH];   // 36 KB
    __shared__ unsigned short B_sh[BN][LDS_PITCH];   // 18 KB

    const int tid   = threadIdx.x;
    const int lane  = tid & 31;
    const int wave  = tid >> 5;
    const int waveM = wave & 3;         // 4 waves along M (64 rows each)
    const int waveN = wave >> 2;        // 2 waves along N (64 cols each)
    const int mlane = lane & 15;
    const int hi    = lane >> 4;        // half-wave selector per ISA layout

    const int mBase = blockIdx.y * BM;
    const int nBase = blockIdx.x * BN;

    // A-tile loader: 16 float4 per row, 16 rows per pass, 16 passes
    const int colA  = (tid & 15) * 4;   // k offset 0..60
    const int rowA0 = tid >> 4;         // 0..15
    // B-tile loader: 8 u16x8 per row, 32 rows per pass, 4 passes
    const int colB  = (tid & 7) * 8;    // k offset 0..56
    const int rowB0 = tid >> 3;         // 0..31

    v8f acc[4][4];
    #pragma unroll
    for (int i = 0; i < 4; ++i)
        #pragma unroll
        for (int j = 0; j < 4; ++j)
            #pragma unroll
            for (int e = 0; e < 8; ++e)
                acc[i][j][e] = 0.0f;

    for (int kt = 0; kt < KDIM; kt += BK) {
        // ---- stage A: x f32 -> bf16 into LDS (256 x 64) ----
        #pragma unroll
        for (int j = 0; j < 16; ++j) {
            const int r = rowA0 + 16 * j;
            f32x4 v = *(const f32x4*)&X[(size_t)(mBase + r) * KDIM + kt + colA];
            u16x4 h;
            h.x = f2bf(v.x); h.y = f2bf(v.y); h.z = f2bf(v.z); h.w = f2bf(v.w);
            *(u16x4*)&A_sh[r][colA] = h;
        }
        // ---- stage B: bf16 weights into LDS (128 x 64) ----
        #pragma unroll
        for (int j = 0; j < 4; ++j) {
            const int r = rowB0 + 32 * j;
            u16x8 v = *(const u16x8*)&W[(size_t)(nBase + r) * KDIM + kt + colB];
            *(u16x8*)&B_sh[r][colB] = v;
        }
        // prefetch next K tile into cache hierarchy (global_prefetch_b8)
        if (kt + BK < KDIM) {
            __builtin_prefetch(&X[(size_t)(mBase + rowA0) * KDIM + kt + BK + colA], 0, 0);
            __builtin_prefetch(&W[(size_t)(nBase + rowB0) * KDIM + kt + BK + colB], 0, 0);
        }
        __syncthreads();

        // ---- compute: 2 k-substeps of 16 WMMAs each ----
        #pragma unroll
        for (int kk = 0; kk < BK; kk += 32) {
            FragBF a[4], b[4];
            // A fragment: lane (m=mlane, hi) holds K chunks [hi*8, +8) and [16+hi*8, +8)
            #pragma unroll
            for (int ms = 0; ms < 4; ++ms) {
                const unsigned short* p = &A_sh[waveM * 64 + ms * 16 + mlane][kk + hi * 8];
                a[ms].h[0] = *(const u16x8*)p;
                a[ms].h[1] = *(const u16x8*)(p + 16);
            }
            // B fragment: identical striping, lane index = N column
            #pragma unroll
            for (int ns = 0; ns < 4; ++ns) {
                const unsigned short* p = &B_sh[waveN * 64 + ns * 16 + mlane][kk + hi * 8];
                b[ns].h[0] = *(const u16x8*)p;
                b[ns].h[1] = *(const u16x8*)(p + 16);
            }
            #pragma unroll
            for (int ms = 0; ms < 4; ++ms)
                #pragma unroll
                for (int ns = 0; ns < 4; ++ns)
                    acc[ms][ns] = __builtin_amdgcn_wmma_f32_16x16x32_bf16(
                        false, a[ms].v, false, b[ns].v,
                        (short)0, acc[ms][ns], false, false);
        }
        __syncthreads();
    }

    // ---- write C: lane = N col, vgpr r = M row (hi picks M>=8) ----
    #pragma unroll
    for (int ms = 0; ms < 4; ++ms) {
        const int m0 = mBase + waveM * 64 + ms * 16 + hi * 8;
        #pragma unroll
        for (int ns = 0; ns < 4; ++ns) {
            const int n0 = nBase + waveN * 64 + ns * 16 + mlane;
            float* po = &out[(size_t)m0 * NDIM + n0];
            #pragma unroll
            for (int r = 0; r < 8; ++r)
                po[(size_t)r * NDIM] = acc[ms][ns][r];
        }
    }
}

// =====================================================================
// Launch
// =====================================================================
extern "C" void kernel_launch(void* const* d_in, const int* in_sizes, int n_in,
                              void* d_out, int out_size, void* d_ws, size_t ws_size,
                              hipStream_t stream) {
    const float*          x           = (const float*)d_in[0];
    const int*            base_packed = (const int*)d_in[1];
    const float*          scales      = (const float*)d_in[2];
    const _Float16*       ortho_vals  = (const _Float16*)d_in[3];
    const int*            ortho_idx   = (const int*)d_in[4];
    float*                out         = (float*)d_out;

    // workspace: bf16 weight matrix [NDIM][KDIM] = 32 MB
    unsigned short* Wbf16 = (unsigned short*)d_ws;

    ortho_dequant_kernel<<<NDIM, 256, 0, stream>>>(
        base_packed, scales, ortho_vals, ortho_idx, Wbf16);

    dim3 grid(NDIM / BN, M_TOT / BM);   // 32 x 32 blocks
    ortho_gemm_kernel<<<grid, 256, 0, stream>>>(x, Wbf16, out);
}
